// Decoder_17300128268670
// MI455X (gfx1250) — matrix operand order repaired and checked by reading, not compiled, not run on hardware
//
#include <hip/hip_runtime.h>
#include <hip/hip_bf16.h>

// ---------------------------------------------------------------------------
// 3-layer LSTM stack for MI455X (gfx1250), wave32 + V_WMMA_F32_16X16X32_F16.
//
//  * One workgroup owns a 16-row batch tile for the WHOLE time loop ->
//    no inter-WG sync; 32 WGs per layer. Recurrence-latency bound problem:
//    ~103 GFLOP total, ~0.25 GB traffic (~11us HBM floor @23.3TB/s), so the
//    goal is minimum per-step critical path.
//  * Layers 1-2: ALL B (weight) fragments hoisted into VGPRs (96 VGPRs) ->
//    zero LDS traffic for weights, no s_wait_dscnt on the B side.
//  * Layer 3: weights resident in LDS (196KB of the 320KB WGP LDS).
//  * x_t tile double-buffered in LDS, staged with GLOBAL_LOAD_ASYNC_TO_LDS
//    (ASYNCcnt) so the DMA of step t+1 overlaps the WMMAs of step t.
//  * c-state in fp32 VGPRs (WMMA C/D layout); h round-trips through LDS f16.
// ---------------------------------------------------------------------------

typedef _Float16 h16;
typedef __attribute__((ext_vector_type(8)))  _Float16 v8h;
typedef __attribute__((ext_vector_type(16))) _Float16 v16h;
typedef __attribute__((ext_vector_type(8)))  float    v8f;
typedef __attribute__((ext_vector_type(4)))  int      v4i;

#if defined(__has_builtin)
#if __has_builtin(__builtin_amdgcn_global_load_async_to_lds_b128) && \
    __has_builtin(__builtin_amdgcn_s_wait_asynccnt)
#define HAVE_ASYNC_LDS 1
#endif
#endif
#ifndef HAVE_ASYNC_LDS
#define HAVE_ASYNC_LDS 0
#endif

typedef __attribute__((address_space(1))) v4i* gv4i_p;
typedef __attribute__((address_space(3))) v4i* lv4i_p;

__device__ __forceinline__ void async_copy16(const h16* g, h16* l) {
#if HAVE_ASYNC_LDS
  __builtin_amdgcn_global_load_async_to_lds_b128((gv4i_p)g, (lv4i_p)l, 0, 0);
#else
  *(uint4*)l = *(const uint4*)g; // sync fallback
#endif
}
__device__ __forceinline__ void async_wait() {
#if HAVE_ASYNC_LDS
  __builtin_amdgcn_s_wait_asynccnt(0);
#endif
}

// A operand (16x32 f16, MxK): lane L holds row M=L&15,
// K = k0 + 16*(L>=16) .. +15 (same k order as loadB).
__device__ __forceinline__ v16h loadA(const h16* tile, int stride, int k0, int lane) {
  const h16* p = tile + (lane & 15) * stride + k0 + ((lane >> 4) << 4);
  return *(const v16h*)p;
}
// B operand (32x16 f16, KxN) stored as W[4H][Kpad] row-major (=B^T):
// lane L holds column N=L&15, K = k0 + 16*(L>=16) .. +15.
__device__ __forceinline__ v16h loadB(const h16* w, int stride, int n, int k0, int lane) {
  return *(const v16h*)(w + (size_t)n * stride + k0 + ((lane >> 4) << 4));
}
__device__ __forceinline__ v8f wmma16(v16h a, v16h b, v8f c) {
  return __builtin_amdgcn_wmma_f32_16x16x32_f16(false, a, false, b, (short)0, c,
                                                false, false);
}
__device__ __forceinline__ float sigm(float x)  { return 1.0f / (1.0f + __expf(-x)); }
__device__ __forceinline__ float tanha(float x) { return 1.0f - 2.0f / (1.0f + __expf(2.0f * x)); }

// ---------------------------------------------------------------------------
// grid.x = B/16.  blockDim.x = (H/16)*32 : one wave per 16-wide hidden
// j-block; each wave computes all 4 gate tiles for its block.
// ---------------------------------------------------------------------------
template <int D, int DPAD, int H, int HPAD, bool HOIST_B, bool LAST>
__global__ __launch_bounds__(256)
void lstm_layer(const h16* __restrict__ x, const h16* __restrict__ wihG,
                const h16* __restrict__ whhG, const float* __restrict__ biasG,
                h16* __restrict__ outH, float* __restrict__ outF,
                float* __restrict__ hnF, int T) {
  constexpr int N4H  = 4 * H;
  constexpr int NW   = H / 16;
  constexpr int NT   = NW * 32;
  constexpr int WLDS = HOIST_B ? 0 : N4H * (DPAD + HPAD); // halves of weight LDS

  extern __shared__ h16 smem[];
  h16* wih = smem;                          // [4H][DPAD]   (only if !HOIST_B)
  h16* whh = smem + (size_t)N4H * DPAD;     // [4H][HPAD]   (only if !HOIST_B)
  h16* xt0 = smem + WLDS;                   // [16][DPAD] double-buffered x_t
  h16* xt1 = xt0 + 16 * DPAD;
  h16* ht  = xt1 + 16 * DPAD;               // [16][HPAD] h state

  const int tid  = threadIdx.x;
  const int lane = tid & 31;
  const int wave = tid >> 5;
  const int b0   = blockIdx.x * 16;
  const int jn   = wave * 16 + (lane & 15); // this lane's hidden column

  // ---- weights: LDS staging (layer 3) -------------------------------------
  if (!HOIST_B) {
    const uint4* s0 = (const uint4*)wihG; uint4* d0 = (uint4*)wih;
    for (int i = tid; i < N4H * DPAD / 8; i += NT) d0[i] = s0[i];
    const uint4* s1 = (const uint4*)whhG; uint4* d1 = (uint4*)whh;
    for (int i = tid; i < N4H * HPAD / 8; i += NT) d1[i] = s1[i];
  }
  // ---- weights: VGPR-resident B fragments (layers 1-2) --------------------
  v16h bI[DPAD / 32][4], bH[HPAD / 32][4];
  if (HOIST_B) {
    #pragma unroll
    for (int kk = 0; kk < DPAD / 32; ++kk)
      #pragma unroll
      for (int g = 0; g < 4; ++g)
        bI[kk][g] = loadB(wihG, DPAD, g * H + jn, kk * 32, lane);
    #pragma unroll
    for (int kk = 0; kk < HPAD / 32; ++kk)
      #pragma unroll
      for (int g = 0; g < 4; ++g)
        bH[kk][g] = loadB(whhG, HPAD, g * H + jn, kk * 32, lane);
  }
  // zero h state and zero-padded K columns of both x buffers
  for (int i = tid; i < 16 * HPAD; i += NT) ht[i] = (h16)0.0f;
  for (int i = tid; i < 2 * 16 * DPAD; i += NT) xt0[i] = (h16)0.0f;
  __syncthreads();

  const float bi = biasG[0 * H + jn];
  const float bf = biasG[1 * H + jn];
  const float bg = biasG[2 * H + jn];
  const float bo = biasG[3 * H + jn];

  const int m_hi = (lane >> 4) << 3; // C/D row offset: +0 / +8
  float cst[8];
  #pragma unroll
  for (int r = 0; r < 8; ++r) cst[r] = 0.0f;

  // async-stage x_t (16 rows x D halves) as 16B chunks into an LDS buffer
  auto stage_x = [&](h16* buf, int t) {
    constexpr int CH = D / 8; // 16B chunks per row
    for (int c = tid; c < 16 * CH; c += NT) {
      int m = c / CH, k8 = (c - m * CH) * 8;
      async_copy16(&x[((size_t)(b0 + m) * T + t) * D + k8], &buf[m * DPAD + k8]);
    }
  };

  stage_x(xt0, 0); // prologue

  for (int t = 0; t < T; ++t) {
    h16* xc = (t & 1) ? xt1 : xt0;
    h16* xn = (t & 1) ? xt0 : xt1;

    async_wait();    // my async chunks for step t have landed in LDS
    __syncthreads(); // barrier A: x(t) + h(t-1) visible to all waves
    if (t + 1 < T) stage_x(xn, t + 1); // DMA for t+1 overlaps step-t WMMAs

    v8f a0 = {bi, bi, bi, bi, bi, bi, bi, bi};
    v8f a1 = {bf, bf, bf, bf, bf, bf, bf, bf};
    v8f a2 = {bg, bg, bg, bg, bg, bg, bg, bg};
    v8f a3 = {bo, bo, bo, bo, bo, bo, bo, bo};

    // g = b + x_t @ Wih^T
    #pragma unroll
    for (int kk = 0; kk < DPAD / 32; ++kk) {
      v16h a = loadA(xc, DPAD, kk * 32, lane);
      if (HOIST_B) {
        a0 = wmma16(a, bI[kk][0], a0); a1 = wmma16(a, bI[kk][1], a1);
        a2 = wmma16(a, bI[kk][2], a2); a3 = wmma16(a, bI[kk][3], a3);
      } else {
        a0 = wmma16(a, loadB(wih, DPAD, 0 * H + jn, kk * 32, lane), a0);
        a1 = wmma16(a, loadB(wih, DPAD, 1 * H + jn, kk * 32, lane), a1);
        a2 = wmma16(a, loadB(wih, DPAD, 2 * H + jn, kk * 32, lane), a2);
        a3 = wmma16(a, loadB(wih, DPAD, 3 * H + jn, kk * 32, lane), a3);
      }
    }
    // g += h_{t-1} @ Whh^T
    #pragma unroll
    for (int kk = 0; kk < HPAD / 32; ++kk) {
      v16h a = loadA(ht, HPAD, kk * 32, lane);
      if (HOIST_B) {
        a0 = wmma16(a, bH[kk][0], a0); a1 = wmma16(a, bH[kk][1], a1);
        a2 = wmma16(a, bH[kk][2], a2); a3 = wmma16(a, bH[kk][3], a3);
      } else {
        a0 = wmma16(a, loadB(whh, HPAD, 0 * H + jn, kk * 32, lane), a0);
        a1 = wmma16(a, loadB(whh, HPAD, 1 * H + jn, kk * 32, lane), a1);
        a2 = wmma16(a, loadB(whh, HPAD, 2 * H + jn, kk * 32, lane), a2);
        a3 = wmma16(a, loadB(whh, HPAD, 3 * H + jn, kk * 32, lane), a3);
      }
    }
    __syncthreads(); // barrier B: all reads of ht done before updating it

    #pragma unroll
    for (int r = 0; r < 8; ++r) {
      float ig = sigm(a0[r]);
      float fg = sigm(a1[r]);
      float gg = tanha(a2[r]);
      float og = sigm(a3[r]);
      float cn = fg * cst[r] + ig * gg;
      cst[r]   = cn;
      float hv = og * tanha(cn);
      int   m  = r + m_hi;
      ht[m * HPAD + jn] = (h16)hv;
      size_t o = ((size_t)(b0 + m) * T + t) * H + jn;
      if (LAST) {
        outF[o] = hv;
        if (t == T - 1) hnF[(size_t)(b0 + m) * H + jn] = hv;
      } else {
        outH[o] = (h16)hv;
      }
    }
  }
}

// ---- prep: f32 -> zero-padded f16 [rows][colspad] --------------------------
__global__ void cvt_pad(const float* __restrict__ s, h16* __restrict__ d,
                        int rows, int cols, int colspad) {
  int i = blockIdx.x * blockDim.x + threadIdx.x;
  if (i >= rows * colspad) return;
  int r = i / colspad, k = i - r * colspad;
  d[i] = (k < cols) ? (h16)s[r * cols + k] : (h16)0.0f;
}
__global__ void bias_add(const float* __restrict__ a, const float* __restrict__ b,
                         float* __restrict__ o, int n) {
  int i = blockIdx.x * blockDim.x + threadIdx.x;
  if (i < n) o[i] = a[i] + b[i];
}

// ---------------------------------------------------------------------------
extern "C" void kernel_launch(void* const* d_in, const int* in_sizes, int n_in,
                              void* d_out, int out_size, void* d_ws, size_t ws_size,
                              hipStream_t stream) {
  (void)in_sizes; (void)n_in; (void)out_size; (void)ws_size;
  const int B = 512, T = 512;

  const float* z    = (const float*)d_in[0];
  const float* Wih1 = (const float*)d_in[1];
  const float* Whh1 = (const float*)d_in[2];
  const float* bih1 = (const float*)d_in[3];
  const float* bhh1 = (const float*)d_in[4];
  const float* Wih2 = (const float*)d_in[5];
  const float* Whh2 = (const float*)d_in[6];
  const float* bih2 = (const float*)d_in[7];
  const float* bhh2 = (const float*)d_in[8];
  const float* Wih3 = (const float*)d_in[9];
  const float* Whh3 = (const float*)d_in[10];
  const float* bih3 = (const float*)d_in[11];
  const float* bhh3 = (const float*)d_in[12];

  // ---- workspace layout (256B aligned) ------------------------------------
  char*  ws  = (char*)d_ws;
  size_t off = 0;
  auto alloc = [&](size_t bytes) -> char* {
    char* p = ws + off;
    off = (off + bytes + 255) & ~(size_t)255;
    return p;
  };
  h16*   w1ih = (h16*)alloc((size_t)128 * 64  * 2);
  h16*   w1hh = (h16*)alloc((size_t)128 * 32  * 2);
  h16*   w2ih = (h16*)alloc((size_t)192 * 32  * 2);
  h16*   w2hh = (h16*)alloc((size_t)192 * 64  * 2);
  h16*   w3ih = (h16*)alloc((size_t)512 * 64  * 2);
  h16*   w3hh = (h16*)alloc((size_t)512 * 128 * 2);
  float* b1   = (float*)alloc(128 * 4);
  float* b2   = (float*)alloc(192 * 4);
  float* b3   = (float*)alloc(512 * 4);
  h16*   z16  = (h16*)alloc((size_t)B * T * 64 * 2); // ~32MB
  h16*   out1 = (h16*)alloc((size_t)B * T * 32 * 2); // ~16MB
  h16*   out2 = (h16*)alloc((size_t)B * T * 48 * 2); // ~24MB

  // ---- weight / bias / input prep -----------------------------------------
  auto grid1 = [](long long n) { return (unsigned)((n + 255) / 256); };
  cvt_pad<<<grid1(128 * 64),  256, 0, stream>>>(Wih1, w1ih, 128, 64,  64);
  cvt_pad<<<grid1(128 * 32),  256, 0, stream>>>(Whh1, w1hh, 128, 32,  32);
  cvt_pad<<<grid1(192 * 32),  256, 0, stream>>>(Wih2, w2ih, 192, 32,  32);
  cvt_pad<<<grid1(192 * 64),  256, 0, stream>>>(Whh2, w2hh, 192, 48,  64);
  cvt_pad<<<grid1(512 * 64),  256, 0, stream>>>(Wih3, w3ih, 512, 48,  64);
  cvt_pad<<<grid1(512 * 128), 256, 0, stream>>>(Whh3, w3hh, 512, 128, 128);
  bias_add<<<grid1(128), 256, 0, stream>>>(bih1, bhh1, b1, 128);
  bias_add<<<grid1(192), 256, 0, stream>>>(bih2, bhh2, b2, 192);
  bias_add<<<grid1(512), 256, 0, stream>>>(bih3, bhh3, b3, 512);
  cvt_pad<<<grid1((long long)B * T * 64), 256, 0, stream>>>(z, z16, B * T, 64, 64);

  // ---- the three recurrent layers -----------------------------------------
  float* dec = (float*)d_out;
  float* hn  = dec + (size_t)B * T * 128;

  const size_t smem1 = (size_t)(2 * 16 * 64 + 16 * 32) * 2;                         // 5 KB
  const size_t smem2 = (size_t)(2 * 16 * 32 + 16 * 64) * 2;                         // 4 KB
  const size_t smem3 = (size_t)(512 * 64 + 512 * 128 + 2 * 16 * 64 + 16 * 128) * 2; // 200 KB

  lstm_layer<64, 64, 32, 32, true, false>
      <<<B / 16, 64, smem1, stream>>>(z16, w1ih, w1hh, b1, out1, nullptr, nullptr, T);
  lstm_layer<32, 32, 48, 64, true, false>
      <<<B / 16, 96, smem2, stream>>>(out1, w2ih, w2hh, b2, out2, nullptr, nullptr, T);
  lstm_layer<48, 64, 128, 128, false, true>
      <<<B / 16, 256, smem3, stream>>>(out2, w3ih, w3hh, b3, nullptr, dec, hn, T);
}
